// LabeledLocalHardAttention_70901320123156
// MI455X (gfx1250) — matrix operand order repaired and checked by reading, not compiled, not run on hardware
//
#include <hip/hip_runtime.h>
#include <hip/hip_bf16.h>

// ---------------------------------------------------------------------------
// LabeledLocalHardAttention for MI455X (gfx1250, wave32, WMMA)
// B=64, S1=S2=256, H=512, L=4
// Pipeline: f32->f16 casts (+one-time weight transpose) -> WMMA GEMM chain
// (MLP x2, score) with DOUBLE-BUFFERED async global->LDS DMA staging ->
// sparse softmax epilogue.
// ---------------------------------------------------------------------------

typedef __attribute__((ext_vector_type(16))) _Float16 v16h;
typedef __attribute__((ext_vector_type(8)))  float    v8f;
typedef __attribute__((ext_vector_type(4)))  unsigned u32x4;

#define LDK 40   // padded LDS K-stride in halves (multiple of 8 -> 16B chunks)

union Frag16 { v16h v; u32x4 q[2]; };

// ---------------------------------------------------------------------------
// CDNA5 async global->LDS DMA (ASYNCcnt-tracked, no VGPR round trip).
// Low 32 bits of a generic pointer to LDS are the wave-relative LDS offset
// (ISA aperture rule).
// ---------------------------------------------------------------------------
__device__ __forceinline__ void async_copy_b128(void* lds_ptr, const void* gptr) {
  unsigned lds_off = (unsigned)(unsigned long long)lds_ptr;
  asm volatile("global_load_async_to_lds_b128 %0, %1, off"
               :: "v"(lds_off), "v"(gptr)
               : "memory");
}
// ASYNCcnt completes in order: <=3 outstanding means the previous batch of 3
// async copies has landed in LDS while the next batch may still be in flight.
__device__ __forceinline__ void wait_async_le3() {
  asm volatile("s_wait_asynccnt 0x3" ::: "memory");
}
__device__ __forceinline__ void wait_async0() {
  asm volatile("s_wait_asynccnt 0x0" ::: "memory");
}

// ---------------------------------------------------------------------------
// f32 -> f16 elementwise convert
// ---------------------------------------------------------------------------
__global__ void cvt_f32_f16_kernel(const float* __restrict__ in,
                                   _Float16* __restrict__ out, long n) {
  long idx    = (long)blockIdx.x * blockDim.x + threadIdx.x;
  long stride = (long)gridDim.x * blockDim.x;
  for (long i = idx; i < n; i += stride) out[i] = (_Float16)in[i];
}

// ---------------------------------------------------------------------------
// f32 [L][K][N] -> f16 [L][N][K] (one-time weight transpose so every GEMM
// stages B with pure 16B vector copies)
// ---------------------------------------------------------------------------
__global__ void cvt_transpose_f32_f16_kernel(const float* __restrict__ in,
                                             _Float16* __restrict__ out,
                                             long n /* = L*H*H */) {
  const int H = 512;
  long idx    = (long)blockIdx.x * blockDim.x + threadIdx.x;
  long stride = (long)gridDim.x * blockDim.x;
  for (long i = idx; i < n; i += stride) {
    long l = i / (H * H);
    long r = i - l * H * H;
    long k = r / H;         // input row (K)
    long nn = r - k * H;    // input col (N)
    out[l * H * H + nn * H + k] = (_Float16)in[i];
  }
}

// ---------------------------------------------------------------------------
// zero fill (d_out is poisoned by the harness; mixed1/mixed2 are sparse)
// ---------------------------------------------------------------------------
__global__ void zero_kernel(float* __restrict__ p, long n) {
  long idx    = (long)blockIdx.x * blockDim.x + threadIdx.x;
  long stride = (long)gridDim.x * blockDim.x;
  for (long i = idx; i < n; i += stride) p[i] = 0.0f;
}

// ---------------------------------------------------------------------------
// Tiled WMMA GEMM:  C[z] = act( A[z](MxK) * Bt[z](NxK)^T + bias[z] )
// B is ALWAYS stored transposed in memory as Bt[N][K] row-major.
//   RELU   : apply relu
//   BIAS   : add f32 bias per output column
//   OUTF16 : write _Float16 (else float)
// Block: 256 threads (8 wave32), block tile 128(M) x 64(N), wave tile 32x32,
// K step 32 via V_WMMA_F32_16X16X32_F16.
// Staging: ping-pong LDS buffers fed by async global->LDS DMA; tile k+1's
// DMA is issued before waiting on tile k, overlapping HBM with the WMMA pipe.
// ---------------------------------------------------------------------------
template <bool RELU, bool BIAS, bool OUTF16>
__global__ __launch_bounds__(256) void gemm_wmma_kernel(
    const _Float16* __restrict__ A, const _Float16* __restrict__ Bt,
    const float* __restrict__ bias, void* __restrict__ Cv,
    int M, int N, int K,
    long aStrideZ, long bStrideZ, long cStrideZ, long biasStrideZ) {
  __shared__ __align__(16) _Float16 As[2][128 * LDK];
  __shared__ __align__(16) _Float16 Bs[2][64 * LDK];

  const int z = blockIdx.z;
  A  += (long)z * aStrideZ;
  Bt += (long)z * bStrideZ;
  const float* biasp = BIAS ? (bias + (long)z * biasStrideZ) : nullptr;

  const int m0 = blockIdx.y * 128;
  const int n0 = blockIdx.x * 64;

  const int t     = threadIdx.x;
  const int lane  = t & 31;
  const int wave  = t >> 5;
  const int waveM = wave >> 1;  // 0..3 -> 32-row slabs
  const int waveN = wave & 1;   // 0..1 -> 32-col slabs
  const int mlo   = lane & 15;
  const int hi    = lane >> 4;

  // per-thread staging assignments (constant across K loop)
  const int aRow0 = t >> 2;                 // 0..63   (A rows, first half)
  const int aRow1 = (t + 256) >> 2;         // 64..127 (A rows, second half)
  const int aCh   = (t & 3) * 8;            // K chunk within 32
  const int bRow  = t >> 2;                 // 0..63   (Bt rows = N)

  auto stage = [&](int kt, int buf) {
    const int k0 = kt << 5;
    async_copy_b128(&As[buf][aRow0 * LDK + aCh],
                    &A[(long)(m0 + aRow0) * K + k0 + aCh]);
    async_copy_b128(&As[buf][aRow1 * LDK + aCh],
                    &A[(long)(m0 + aRow1) * K + k0 + aCh]);
    async_copy_b128(&Bs[buf][bRow * LDK + aCh],
                    &Bt[(long)(n0 + bRow) * K + k0 + aCh]);
  };

  v8f acc[2][2] = {};

  const int NK = K >> 5;  // K-steps of 32
  stage(0, 0);            // prologue: DMA tile 0 into buffer 0

  for (int kt = 0; kt < NK; ++kt) {
    const int cur = kt & 1;
    // all waves finished reading buf[cur^1] (iteration kt-1) -> safe to refill
    __syncthreads();
    if (kt + 1 < NK) {
      stage(kt + 1, cur ^ 1);  // overlap next tile's DMA with this tile's math
      wait_async_le3();        // tile kt's 3 copies landed (in-order ASYNCcnt)
    } else {
      wait_async0();
    }
    __syncthreads();           // all waves' buf[cur] data visible

    // ---- load fragments per ISA VGPR layouts ----
    // A frag: lane m=lane&15; halves K = hi*8..+7 and 16+hi*8..+7
    // B frag: lane n=lane&15; halves K = hi*8..+7 and 16+hi*8..+7
    Frag16 fa[2], fb[2];
#pragma unroll
    for (int mt = 0; mt < 2; ++mt) {
      int row = waveM * 32 + mt * 16 + mlo;
      fa[mt].q[0] = *(const u32x4*)&As[cur][row * LDK + hi * 8];
      fa[mt].q[1] = *(const u32x4*)&As[cur][row * LDK + 16 + hi * 8];
    }
#pragma unroll
    for (int nt = 0; nt < 2; ++nt) {
      int row = waveN * 32 + nt * 16 + mlo;
      fb[nt].q[0] = *(const u32x4*)&Bs[cur][row * LDK + hi * 8];
      fb[nt].q[1] = *(const u32x4*)&Bs[cur][row * LDK + 16 + hi * 8];
    }
#pragma unroll
    for (int mt = 0; mt < 2; ++mt)
#pragma unroll
      for (int nt = 0; nt < 2; ++nt)
        acc[mt][nt] = __builtin_amdgcn_wmma_f32_16x16x32_f16(
            false, fa[mt].v, false, fb[nt].v, (short)0, acc[mt][nt], false,
            false);
  }

  // ---- epilogue: C/D layout -> m = v + hi*8, n = lane&15 ----
  // one 64-bit base per 16x16 tile; row v is stored at offset v*N (N is a runtime arg).
#pragma unroll
  for (int mt = 0; mt < 2; ++mt) {
#pragma unroll
    for (int nt = 0; nt < 2; ++nt) {
      const int n = n0 + waveN * 32 + nt * 16 + mlo;
      const int mbase = m0 + waveM * 32 + mt * 16 + hi * 8;
      const float bval = BIAS ? biasp[n] : 0.0f;
      const long base = (long)z * cStrideZ + (long)mbase * N + n;
      if (OUTF16) {
        _Float16* cp = (_Float16*)Cv + base;
#pragma unroll
        for (int v = 0; v < 8; ++v) {
          float val = acc[mt][nt][v] + bval;
          if (RELU) val = val > 0.0f ? val : 0.0f;
          cp[v * N] = (_Float16)val;
        }
      } else {
        float* cp = (float*)Cv + base;
#pragma unroll
        for (int v = 0; v < 8; ++v) {
          float val = acc[mt][nt][v] + bval;
          if (RELU) val = val > 0.0f ? val : 0.0f;
          cp[v * N] = val;
        }
      }
    }
  }
}

// ---------------------------------------------------------------------------
// Row stats over score1[b,i, l*256+j] = score[l,b,i,j]:
//   inv Z1 (softmax denom after max shift) and argmax index k = l*256+j.
// One block per (b,i); 256 threads x 4 elements.
// ---------------------------------------------------------------------------
__global__ __launch_bounds__(256) void rowstats1_kernel(
    const float* __restrict__ score, float* __restrict__ invZ1,
    int* __restrict__ amax1) {
  const int bi = blockIdx.x;  // b*256 + i
  const int b = bi >> 8, i = bi & 255;
  const int t = threadIdx.x;
  __shared__ float sm[256];
  __shared__ int   sa[256];

  float vals[4];
  float m = -3.4e38f;
  int   a = 0x7fffffff;
#pragma unroll
  for (int r = 0; r < 4; ++r) {
    int k = t + r * 256;
    int l = k >> 8, j = k & 255;
    vals[r] = score[(((long)l * 64 + b) * 256 + i) * 256 + j];
    if (vals[r] > m || (vals[r] == m && k < a)) { m = vals[r]; a = k; }
  }
  sm[t] = m; sa[t] = a;
  __syncthreads();
  for (int s = 128; s > 0; s >>= 1) {
    if (t < s) {
      if (sm[t + s] > sm[t] || (sm[t + s] == sm[t] && sa[t + s] < sa[t])) {
        sm[t] = sm[t + s]; sa[t] = sa[t + s];
      }
    }
    __syncthreads();
  }
  const float gmax = sm[0];
  const int   gidx = sa[0];
  __syncthreads();

  float e = 0.0f;
#pragma unroll
  for (int r = 0; r < 4; ++r) e += __expf(vals[r] - gmax);
  sm[t] = e;
  __syncthreads();
  for (int s = 128; s > 0; s >>= 1) {
    if (t < s) sm[t] += sm[t + s];
    __syncthreads();
  }
  if (t == 0) { invZ1[bi] = 1.0f / sm[0]; amax1[bi] = gidx; }
}

// ---------------------------------------------------------------------------
// Row stats over score2[b,j, l*256+i] = score[l,b,i,j]: max2 and inv Z2.
// ---------------------------------------------------------------------------
__global__ __launch_bounds__(256) void rowstats2_kernel(
    const float* __restrict__ score, float* __restrict__ max2,
    float* __restrict__ invZ2) {
  const int bj = blockIdx.x;  // b*256 + j
  const int b = bj >> 8, j = bj & 255;
  const int t = threadIdx.x;
  __shared__ float sm[256];

  float vals[4];
  float m = -3.4e38f;
#pragma unroll
  for (int r = 0; r < 4; ++r) {
    int k = t + r * 256;  // k = l*256 + i
    int l = k >> 8, i = k & 255;
    vals[r] = score[(((long)l * 64 + b) * 256 + i) * 256 + j];
    m = vals[r] > m ? vals[r] : m;
  }
  sm[t] = m;
  __syncthreads();
  for (int s = 128; s > 0; s >>= 1) {
    if (t < s) sm[t] = sm[t + s] > sm[t] ? sm[t + s] : sm[t];
    __syncthreads();
  }
  const float gmax = sm[0];
  __syncthreads();
  float e = 0.0f;
#pragma unroll
  for (int r = 0; r < 4; ++r) e += __expf(vals[r] - gmax);
  sm[t] = e;
  __syncthreads();
  for (int s = 128; s > 0; s >>= 1) {
    if (t < s) sm[t] += sm[t + s];
    __syncthreads();
  }
  if (t == 0) { max2[bj] = gmax; invZ2[bj] = 1.0f / sm[0]; }
}

// ---------------------------------------------------------------------------
// Sparse scatter: mixed1 has one nonzero per (b,i) at k*=argmax (value 1/Z1);
// mixed2's nonzeros are exactly those positions transposed per label:
//   mixed2[b, j*, l* *256 + i] = exp(score[l*,b,i,j*] - max2[b,j*]) / Z2[b,j*]
// ---------------------------------------------------------------------------
__global__ void scatter_kernel(const float* __restrict__ score,
                               const float* __restrict__ invZ1,
                               const int* __restrict__ amax1,
                               const float* __restrict__ max2,
                               const float* __restrict__ invZ2,
                               float* __restrict__ out) {
  const int t = blockIdx.x * blockDim.x + threadIdx.x;  // 0..16383 = b*256+i
  const int b = t >> 8, i = t & 255;
  const int k = amax1[t];
  const int l = k >> 8, j = k & 255;

  out[((long)(b * 256 + i)) * 1024 + k] = invZ1[t];

  const float s  = score[(((long)l * 64 + b) * 256 + i) * 256 + j];
  const int   bj = b * 256 + j;
  const long  M1 = (long)64 * 256 * 1024;
  out[M1 + (long)bj * 1024 + l * 256 + i] = __expf(s - max2[bj]) * invZ2[bj];
}

// ---------------------------------------------------------------------------
// Host launcher
// ---------------------------------------------------------------------------
extern "C" void kernel_launch(void* const* d_in, const int* in_sizes, int n_in,
                              void* d_out, int out_size, void* d_ws,
                              size_t ws_size, hipStream_t stream) {
  (void)in_sizes; (void)n_in; (void)ws_size;
  const float* sent1 = (const float*)d_in[0];  // [64,256,512]
  const float* sent2 = (const float*)d_in[1];  // [64,256,512]
  const float* W1    = (const float*)d_in[2];  // [4,512,512]
  const float* b1    = (const float*)d_in[3];  // [4,512]
  const float* W2    = (const float*)d_in[4];  // [4,512,512]
  const float* b2    = (const float*)d_in[5];  // [4,512]
  float* out = (float*)d_out;                  // [2, 64,256,1024] concat

  const long M = 64L * 256;  // 16384 rows per sentence
  const int  H = 512;
  const int  L = 4;

  // carve workspace
  char*  ws  = (char*)d_ws;
  size_t off = 0;
  auto carve = [&](size_t bytes) {
    void* p = ws + off;
    off += (bytes + 255) & ~(size_t)255;
    return p;
  };
  _Float16* x1h = (_Float16*)carve(M * H * 2);             // 16 MB
  _Float16* x2h = (_Float16*)carve(M * H * 2);             // 16 MB
  _Float16* w1t = (_Float16*)carve((size_t)L * H * H * 2); // 2 MB  [L][N][K]
  _Float16* w2t = (_Float16*)carve((size_t)L * H * H * 2); // 2 MB  [L][N][K]
  _Float16* h1  = (_Float16*)carve((size_t)L * M * H * 2); // 64 MB
  _Float16* h2  = (_Float16*)carve((size_t)L * M * H * 2); // 64 MB
  _Float16* g1  = (_Float16*)carve((size_t)L * M * H * 2); // 64 MB
  _Float16* g2  = (_Float16*)carve((size_t)L * M * H * 2); // 64 MB
  float* score  = (float*)carve((size_t)L * 64 * 256 * 256 * 4);  // 64 MB
  float* invZ1  = (float*)carve(16384 * 4);
  int*   amax1  = (int*)carve(16384 * 4);
  float* max2   = (float*)carve(16384 * 4);
  float* invZ2  = (float*)carve(16384 * 4);

  // 1) precision cast to f16; weights also transposed to [L][N][K]
  cvt_f32_f16_kernel<<<1024, 256, 0, stream>>>(sent1, x1h, M * H);
  cvt_f32_f16_kernel<<<1024, 256, 0, stream>>>(sent2, x2h, M * H);
  cvt_transpose_f32_f16_kernel<<<512, 256, 0, stream>>>(W1, w1t,
                                                        (long)L * H * H);
  cvt_transpose_f32_f16_kernel<<<512, 256, 0, stream>>>(W2, w2t,
                                                        (long)L * H * H);

  // 2) zero the sparse outputs (harness poisons d_out)
  zero_kernel<<<2048, 256, 0, stream>>>(out, (long)out_size);

  // 3) MLP layer 1: h = relu(x @ W1[l] + b1[l]) for both sentences
  dim3 mlpGrid(H / 64, M / 128, L);  // (8,128,4)
  gemm_wmma_kernel<true, true, true><<<mlpGrid, 256, 0, stream>>>(
      x1h, w1t, b1, h1, (int)M, H, H, 0L, (long)H * H, M * H, (long)H);
  gemm_wmma_kernel<true, true, true><<<mlpGrid, 256, 0, stream>>>(
      x2h, w1t, b1, h2, (int)M, H, H, 0L, (long)H * H, M * H, (long)H);

  // 4) MLP layer 2: g = relu(h @ W2[l] + b2[l])
  gemm_wmma_kernel<true, true, true><<<mlpGrid, 256, 0, stream>>>(
      h1, w2t, b2, g1, (int)M, H, H, M * H, (long)H * H, M * H, (long)H);
  gemm_wmma_kernel<true, true, true><<<mlpGrid, 256, 0, stream>>>(
      h2, w2t, b2, g2, (int)M, H, H, M * H, (long)H * H, M * H, (long)H);

  // 5) score[l,b] = g1[l,b] @ g2[l,b]^T   (z = l*64 + b; g2 rows are Bt[N][K])
  dim3 scoreGrid(256 / 64, 256 / 128, L * 64);  // (4,2,256)
  gemm_wmma_kernel<false, false, false><<<scoreGrid, 256, 0, stream>>>(
      g1, g2, nullptr, score, 256, 256, H, 256L * H, 256L * H, 256L * 256, 0L);

  // 6) softmax/argmax row stats over both joint axes (1024 = L*S)
  rowstats1_kernel<<<16384, 256, 0, stream>>>(score, invZ1, amax1);
  rowstats2_kernel<<<16384, 256, 0, stream>>>(score, max2, invZ2);

  // 7) sparse scatter of mixed1/mixed2
  scatter_kernel<<<64, 256, 0, stream>>>(score, invZ1, amax1, max2, invZ2, out);
}
